// WaveFormer_69793218560504
// MI455X (gfx1250) — compile-verified, matrix operand order checked by reading, not run on hardware
//
#include <hip/hip_runtime.h>
#include <hip/hip_bf16.h>

typedef __attribute__((ext_vector_type(16))) _Float16 v16h;
typedef __attribute__((ext_vector_type(8)))  float    v8f;
typedef __attribute__((ext_vector_type(2)))  float    v2f;

#define HWW 3136   // 56*56
#define CH  384
#define BB  16
#define MPIX (BB*HWW)   // 50176
#define PI_F 3.14159265358979323846f

#if defined(__HIP_DEVICE_COMPILE__) && __has_builtin(__builtin_amdgcn_wmma_f32_16x16x4_f32)
#define HAVE_WMMA_F32 1
#else
#define HAVE_WMMA_F32 0
#endif

// ---------------- small init / conversion kernels ----------------

__global__ void init_dct_kernel(float* __restrict__ D, float* __restrict__ decay) {
    int i = blockIdx.x * 256 + threadIdx.x;
    if (i >= 56 * 56) return;
    int k = i / 56, n = i % 56;
    float v = cosf(PI_F * (2.0f * n + 1.0f) * k / 112.0f) * sqrtf(2.0f / 56.0f);
    if (k == 0) v *= 0.70710678118654752f;
    D[i] = v;
    float wn = PI_F * (float)k / 56.0f;
    float wm = PI_F * (float)n / 56.0f;
    decay[i] = expf(-(wn * wn + wm * wm));
}

__global__ void f32_to_f16_kernel(const float* __restrict__ src, _Float16* __restrict__ dst, int n) {
    int i = blockIdx.x * 256 + threadIdx.x;
    if (i < n) dst[i] = (_Float16)src[i];
}

// ---------------- depthwise 3x3 conv (NCHW in) -> pixel-major f16 [M, C] ----------------

__global__ __launch_bounds__(256)
void dwconv_kernel(const float* __restrict__ x, const float* __restrict__ w,
                   const float* __restrict__ bias, _Float16* __restrict__ xlh) {
    long i = (long)blockIdx.x * 256 + threadIdx.x;
    const long total = (long)BB * CH * HWW;
    if (i >= total) return;
    int hw = (int)(i % HWW);
    long t = i / HWW;
    int c = (int)(t % CH);
    int b = (int)(t / CH);
    int h = hw / 56, ww = hw % 56;
    const float* xp = x + ((long)(b * CH + c)) * HWW;
    const float* wp = w + c * 9;
    float s = bias[c];
#pragma unroll
    for (int dy = -1; dy <= 1; ++dy) {
        int hh = h + dy;
        if (hh < 0 || hh >= 56) continue;
#pragma unroll
        for (int dx = -1; dx <= 1; ++dx) {
            int wv = ww + dx;
            if (wv < 0 || wv >= 56) continue;
            s += xp[hh * 56 + wv] * wp[(dy + 1) * 3 + (dx + 1)];
        }
    }
    xlh[((long)(b * HWW + hw)) * CH + c] = (_Float16)s;
}

// ---------------- WMMA GEMM: D[M,N] = A[M,K] @ W[N,K]^T + bias ----------------
// One wave computes a 16x64 output strip (4 n-tiles), reusing the A fragment 4x.
// Fragments packed per CDNA5 ISA 16-bit A 16x32 layout; fp32 accumulate.

enum { MLIN = 0, MVEL = 1, MTOK = 2, MOUT = 3 };

template <int MODE>
__global__ __launch_bounds__(256)
void gemm16x4_kernel(const _Float16* __restrict__ A, const _Float16* __restrict__ Wm,
                     const float* __restrict__ bias, int M, int N, int K,
                     _Float16* __restrict__ o_h, _Float16* __restrict__ o_h2,
                     float* __restrict__ o_f, float* __restrict__ o_f2,
                     const float* __restrict__ cptr) {
    int lane = threadIdx.x & 31;
    int wave = threadIdx.x >> 5;
    int ng = N >> 6;                       // groups of 4 n-tiles
    long unit = (long)blockIdx.x * 8 + wave;
    long total = (long)(M >> 4) * ng;
    if (unit >= total) return;
    int tm = (int)(unit / ng);
    int g  = (int)(unit % ng);
    int r = lane & 15;
    int hh = lane >> 4;                    // lane half selects K sub-block per ISA layout

    const _Float16* Ap  = A  + ((long)(tm * 16 + r)) * K;
    const _Float16* Wp0 = Wm + ((long)(g * 64 + r)) * K;

    v8f acc[4] = {{}, {}, {}, {}};
    for (int k0 = 0; k0 < K; k0 += 32) {
        v16h afr;
#pragma unroll
        for (int v = 0; v < 8; ++v) {
            // VGPR v pair covers K = (v<4?0:16) + (v%4)*2 + half*8
            int klo = k0 + ((v & 3) << 1) + (hh << 3) + ((v & 4) << 2);
            afr[2 * v]     = Ap[klo];
            afr[2 * v + 1] = Ap[klo + 1];
        }
#pragma unroll
        for (int nt = 0; nt < 4; ++nt) {
            const _Float16* Wp = Wp0 + (long)nt * 16 * K;
            v16h bfr;
#pragma unroll
            for (int v = 0; v < 8; ++v) {
                int klo = k0 + ((v & 3) << 1) + (hh << 3) + ((v & 4) << 2);
                bfr[2 * v]     = Wp[klo];
                bfr[2 * v + 1] = Wp[klo + 1];
            }
            acc[nt] = __builtin_amdgcn_wmma_f32_16x16x32_f16(
                false, afr, false, bfr, (short)0, acc[nt], false, false);
        }
    }

    float cs = 0.0f;
    if constexpr (MODE == MTOK) cs = cptr[0];

#pragma unroll
    for (int nt = 0; nt < 4; ++nt) {
        int coln = g * 64 + nt * 16 + r;
        float bv = bias[coln];
#pragma unroll
        for (int j = 0; j < 8; ++j) {
            int row = tm * 16 + j + 8 * hh;    // M index per C/D VGPR layout
            float val = acc[nt][j] + bv;
            if constexpr (MODE == MLIN) {
                int b = row / HWW, hw = row % HWW;
                if (coln < CH) {
                    o_h[(long)row * CH + coln] = (_Float16)val;                    // xg pixel-major f16
                    o_f[((long)b * CH + coln) * HWW + hw] = val;                   // xg channel-major f32
                } else {
                    o_h2[((long)b * CH + (coln - CH)) * HWW + hw] = (_Float16)val; // z channel-major f16
                }
            } else if constexpr (MODE == MVEL) {
                int b = row / HWW, hw = row % HWW;
                o_f[((long)b * CH + coln) * HWW + hw] = val;
            } else if constexpr (MODE == MTOK) {
                float t = val > 0.0f ? val : 0.0f;
                o_f [(long)coln * HWW + row] = cosf(cs * t);
                o_f2[(long)coln * HWW + row] = sinf(cs * t) / (cs + 1e-6f);
            } else { // MOUT
                int b = row / HWW, hw = row % HWW;
                o_f[((long)b * CH + coln) * HWW + hw] = val;
            }
        }
    }
}

// ---------------- 56x56 LDS matmul on the matrix pipes (f32 WMMA, exact) ----------------
// O = op(A) @ op(B), all matrices 56x56 row-major in LDS (O may be global).
// TA: A element A[m,k] read as A56[k*56+m] (i.e. A^T); TB: B element B[k,n] read as B56[n*56+k].
// Pads 56 -> 64 tiles; out-of-range LDS reads return 0 / touch discarded rows-cols; stores masked.

template <bool TA, bool TB>
__device__ __forceinline__
void mm56(const float* A56, const float* B56, float* O56,
          const float* __restrict__ scale, int wave, int lane) {
    int r = lane & 15;
    int hh = lane >> 4;
    for (int t = wave; t < 16; t += 8) {
        int tm = t >> 2, tn = t & 3;
        int m = tm * 16 + r;
        int n = tn * 16 + r;
#if HAVE_WMMA_F32
        v8f acc = {};
        for (int k0 = 0; k0 < 56; k0 += 4) {
            int k = k0 + 2 * hh;           // f32 A 16x4 layout: lanes 0-15 K=0,1; lanes 16-31 K=2,3
            v2f a, b;
            a[0] = TA ? A56[(k + 0) * 56 + m] : A56[m * 56 + (k + 0)];
            a[1] = TA ? A56[(k + 1) * 56 + m] : A56[m * 56 + (k + 1)];
            b[0] = TB ? B56[n * 56 + (k + 0)] : B56[(k + 0) * 56 + n];
            b[1] = TB ? B56[n * 56 + (k + 1)] : B56[(k + 1) * 56 + n];
            acc = __builtin_amdgcn_wmma_f32_16x16x4_f32(
                false, a, false, b, (short)0, acc, false, false);
        }
        if (n < 56) {
#pragma unroll
            for (int j = 0; j < 8; ++j) {
                int row = tm * 16 + j + 8 * hh;
                if (row < 56) {
                    float v = acc[j];
                    if (scale) v *= scale[row * 56 + n];
                    O56[row * 56 + n] = v;
                }
            }
        }
#else
        // VALU fallback (host pass / missing builtin)
        for (int jj = 0; jj < 8; ++jj) {
            int row = tm * 16 + jj + 8 * hh;
            if (row < 56 && n < 56) {
                float s = 0.0f;
                for (int k = 0; k < 56; ++k) {
                    float av = TA ? A56[k * 56 + row] : A56[row * 56 + k];
                    float bv = TB ? B56[n * 56 + k]  : B56[k * 56 + n];
                    s += av * bv;
                }
                if (scale) s *= scale[row * 56 + n];
                O56[row * 56 + n] = s;
            }
        }
#endif
    }
}

// ---------------- fused DCT2D -> decay -> modulation -> IDCT2D per (b,c) 56x56 tile ----------------

__global__ __launch_bounds__(256)
void dct_mod_kernel(const float* __restrict__ D, const float* __restrict__ decay,
                    float* __restrict__ xu_cm, const float* __restrict__ xv_cm,
                    const float* __restrict__ cosm, const float* __restrict__ sinm,
                    const float* __restrict__ alpha_p) {
    __shared__ float Ds[3136];
    __shared__ float Xu[3136];
    __shared__ float Xv[3136];
    __shared__ float T [3136];

    int bc = blockIdx.x;            // b*CH + c
    int c  = bc % CH;
    long base = (long)bc * HWW;
    int lane = threadIdx.x & 31;
    int wave = threadIdx.x >> 5;

    for (int i = threadIdx.x; i < 3136; i += 256) {
        Ds[i] = D[i];
        Xu[i] = xu_cm[base + i];
        Xv[i] = xv_cm[base + i];
    }
    __syncthreads();

    // T = D @ Xu
    mm56<false, false>(Ds, Xu, T, nullptr, wave, lane);
    __syncthreads();
    // Xu = (T @ D^T) * decay
    mm56<false, true>(T, Ds, Xu, decay, wave, lane);
    __syncthreads();
    // T = D @ Xv
    mm56<false, false>(Ds, Xv, T, nullptr, wave, lane);
    __syncthreads();
    // Xv = (T @ D^T) * decay
    mm56<false, true>(T, Ds, Xv, decay, wave, lane);
    __syncthreads();

    // modulation: T = cos*Xu + sin*(Xv + 0.5*alpha*Xu)
    float alpha = alpha_p[0];
    for (int i = threadIdx.x; i < 3136; i += 256) {
        float ct = cosm[(long)c * HWW + i];
        float st = sinm[(long)c * HWW + i];
        T[i] = ct * Xu[i] + st * (Xv[i] + 0.5f * alpha * Xu[i]);
    }
    __syncthreads();

    // Xu = D^T @ T
    mm56<true, false>(Ds, T, Xu, nullptr, wave, lane);
    __syncthreads();
    // y = Xu @ D  -> overwrite this (b,c) tile of xu_cm
    mm56<false, false>(Xu, Ds, xu_cm + base, nullptr, wave, lane);
}

// ---------------- LayerNorm over C + SiLU gate -> pixel-major f16 [M, C] ----------------

__global__ __launch_bounds__(256)
void ln_gate_kernel(const float* __restrict__ y_cm, const _Float16* __restrict__ z_cm,
                    const float* __restrict__ gam, const float* __restrict__ bet,
                    _Float16* __restrict__ gated) {
    int pix = blockIdx.x * 256 + threadIdx.x;
    if (pix >= MPIX) return;
    int b = pix / HWW, hw = pix % HWW;
    const float*    yb = y_cm + (long)b * CH * HWW + hw;
    const _Float16* zb = z_cm + (long)b * CH * HWW + hw;
    float s = 0.0f, ss = 0.0f;
    for (int c = 0; c < CH; ++c) {
        float v = yb[(long)c * HWW];
        s += v; ss += v * v;
    }
    float mu = s * (1.0f / CH);
    float var = ss * (1.0f / CH) - mu * mu;
    float inv = rsqrtf(var + 1e-5f);
    for (int c = 0; c < CH; ++c) {
        float v  = (yb[(long)c * HWW] - mu) * inv * gam[c] + bet[c];
        float zv = (float)zb[(long)c * HWW];
        float sz = zv / (1.0f + expf(-zv));   // silu
        gated[(long)pix * CH + c] = (_Float16)(v * sz);
    }
}

// ---------------- launch ----------------

extern "C" void kernel_launch(void* const* d_in, const int* in_sizes, int n_in,
                              void* d_out, int out_size, void* d_ws, size_t ws_size,
                              hipStream_t stream) {
    (void)in_sizes; (void)n_in; (void)out_size; (void)ws_size;

    const float* x      = (const float*)d_in[0];
    const float* femb   = (const float*)d_in[1];
    const float* dw_w   = (const float*)d_in[2];
    const float* dw_b   = (const float*)d_in[3];
    const float* lin_w  = (const float*)d_in[4];
    const float* lin_b  = (const float*)d_in[5];
    const float* vel_w  = (const float*)d_in[6];
    const float* vel_b  = (const float*)d_in[7];
    const float* tok_w  = (const float*)d_in[8];
    const float* tok_b  = (const float*)d_in[9];
    const float* ln_g   = (const float*)d_in[10];
    const float* ln_b   = (const float*)d_in[11];
    const float* out_w  = (const float*)d_in[12];
    const float* out_b  = (const float*)d_in[13];
    const float* cS     = (const float*)d_in[14];
    const float* alphaS = (const float*)d_in[15];
    float* out = (float*)d_out;

    char* ws = (char*)d_ws;
    size_t off = 0;
    auto take = [&](size_t bytes) -> char* {
        char* p = ws + off;
        off += (bytes + 255) & ~(size_t)255;
        return p;
    };

    float*    Dm     = (float*)take(3136 * 4);
    float*    decay  = (float*)take(3136 * 4);
    _Float16* linWh  = (_Float16*)take((size_t)768 * 384 * 2);
    _Float16* velWh  = (_Float16*)take((size_t)384 * 384 * 2);
    _Float16* tokWh  = (_Float16*)take((size_t)384 * 384 * 2);
    _Float16* outWh  = (_Float16*)take((size_t)384 * 384 * 2);
    _Float16* fembH  = (_Float16*)take((size_t)HWW * CH * 2);
    float*    cosm   = (float*)take((size_t)CH * HWW * 4);
    float*    sinm   = (float*)take((size_t)CH * HWW * 4);
    _Float16* xlH    = (_Float16*)take((size_t)MPIX * CH * 2);  // conv out; reused as gated
    _Float16* xgH    = (_Float16*)take((size_t)MPIX * CH * 2);
    _Float16* zH     = (_Float16*)take((size_t)MPIX * CH * 2);  // channel-major [B,C,HW]
    float*    xuCM   = (float*)take((size_t)MPIX * CH * 4);     // xg channel-major; reused as y
    float*    xvCM   = (float*)take((size_t)MPIX * CH * 4);

    // init + weight conversions
    init_dct_kernel<<<(3136 + 255) / 256, 256, 0, stream>>>(Dm, decay);
    f32_to_f16_kernel<<<(768 * 384 + 255) / 256, 256, 0, stream>>>(lin_w, linWh, 768 * 384);
    f32_to_f16_kernel<<<(384 * 384 + 255) / 256, 256, 0, stream>>>(vel_w, velWh, 384 * 384);
    f32_to_f16_kernel<<<(384 * 384 + 255) / 256, 256, 0, stream>>>(tok_w, tokWh, 384 * 384);
    f32_to_f16_kernel<<<(384 * 384 + 255) / 256, 256, 0, stream>>>(out_w, outWh, 384 * 384);
    f32_to_f16_kernel<<<(HWW * CH + 255) / 256, 256, 0, stream>>>(femb, fembH, HWW * CH);

    // depthwise conv -> xl (pixel-major f16)
    dwconv_kernel<<<(int)(((long)BB * CH * HWW + 255) / 256), 256, 0, stream>>>(x, dw_w, dw_b, xlH);

    // lin GEMM: xz = xl @ lin_w^T + lin_b ; split into xg (f16 + channel-major f32) and z
    gemm16x4_kernel<MLIN><<<(MPIX / 16) * (768 / 64) / 8, 256, 0, stream>>>(
        xlH, linWh, lin_b, MPIX, 768, CH, xgH, zH, xuCM, nullptr, nullptr);

    // vel GEMM: x_vel = xg @ vel_w^T + vel_b (channel-major f32)
    gemm16x4_kernel<MVEL><<<(MPIX / 16) * (CH / 64) / 8, 256, 0, stream>>>(
        xgH, velWh, vel_b, MPIX, CH, CH, nullptr, nullptr, xvCM, nullptr, nullptr);

    // tok GEMM: t = relu(femb @ tok_w^T + tok_b) -> cos/sin maps [C,HW]
    gemm16x4_kernel<MTOK><<<(HWW / 16) * (CH / 64) / 8, 256, 0, stream>>>(
        fembH, tokWh, tok_b, HWW, CH, CH, nullptr, nullptr, cosm, sinm, cS);

    // fused DCT2D -> decay -> modulation -> IDCT2D (y written back into xuCM)
    dct_mod_kernel<<<BB * CH, 256, 0, stream>>>(Dm, decay, xuCM, xvCM, cosm, sinm, alphaS);

    // LayerNorm + SiLU gate -> gated (reuse xlH)
    ln_gate_kernel<<<(MPIX + 255) / 256, 256, 0, stream>>>(xuCM, zH, ln_g, ln_b, xlH);

    // out GEMM: out = gated @ out_w^T + out_b -> NCHW output
    gemm16x4_kernel<MOUT><<<(MPIX / 16) * (CH / 64) / 8, 256, 0, stream>>>(
        xlH, outWh, out_b, MPIX, CH, CH, nullptr, nullptr, out, nullptr, nullptr);
}
